// R_GCN_35476429865087
// MI455X (gfx1250) — matrix operand order repaired and checked by reading, not compiled, not run on hardware
//
#include <hip/hip_runtime.h>
#include <hip/hip_bf16.h>
#include <cstdint>
#include <cstddef>

// Problem dimensions (match reference)
#define NPTS   40000
#define KPNB   16
#define GEO    8
#define NCELL  100000
#define NFACET 200000
#define EPSBN  1e-6f

typedef __attribute__((ext_vector_type(16))) _Float16 v16h;
typedef __attribute__((ext_vector_type(8)))  float    v8f;

enum { ACT_NONE = 0, ACT_RELU = 1, ACT_L02 = 2, ACT_L01 = 3 };

__device__ __forceinline__ float act_apply(float x, int act) {
  switch (act) {
    case ACT_RELU: return x > 0.f ? x : 0.f;
    case ACT_L02:  return x >= 0.f ? x : 0.02f * x;
    case ACT_L01:  return x >= 0.f ? x : 0.01f * x;
    default:       return x;
  }
}

// ---------------------------------------------------------------------------
// WMMA GEMM:  C[N,dout] = A[N,din] * W[dout,din]^T + bias
// Block: 256 threads = 8 waves; tile M=128 x N=32, K-step 32.
// 8 waves = 4 M-groups (32 rows) x 2 N-subtiles; 2 independent
// v_wmma_f32_16x16x32_f16 per wave per K-step (shared B fragment).
// Double-buffered LDS (ping-pong): next tile's global loads overlap the
// current tile's WMMAs; one barrier per K-step. KFULL: din%32==0 (no k-edge
// predicates). CHECKM selected per-block by a uniform scalar branch so
// interior blocks carry no row predicates at all.
// ---------------------------------------------------------------------------
#define TM 128
#define TN 32
#define TK 32

union FragH  { v16h v; uint4 q[2]; _Float16 h[16]; };
union Pack16 { uint4 q[2]; _Float16 h[16]; };
union Pack4  { uint2 q;    _Float16 h[4];  };
union AccU   { v8f v; float f[8]; };

template <bool KFULL>
__device__ __forceinline__ void fetch_a(float* f, const float* Ap0, bool rowOk,
                                        int din, int k0, int aK)
{
  const float* Ap = Ap0 + k0;
  if (KFULL) {
    if (rowOk) {
      #pragma unroll
      for (int i = 0; i < 16; ++i) f[i] = Ap[i];
    } else {
      #pragma unroll
      for (int i = 0; i < 16; ++i) f[i] = 0.f;
    }
  } else {
    const int rem = din - k0 - aK;
    if (rowOk && rem >= 16) {
      #pragma unroll
      for (int i = 0; i < 16; ++i) f[i] = Ap[i];
    } else if (rowOk && rem > 0) {
      #pragma unroll
      for (int i = 0; i < 16; ++i) f[i] = (i < rem) ? Ap[i] : 0.f;
    } else {
      #pragma unroll
      for (int i = 0; i < 16; ++i) f[i] = 0.f;
    }
  }
}

template <bool KFULL>
__device__ __forceinline__ void fetch_b(float* f, const float* Wp0, bool rowOk,
                                        int din, int k0, int bK)
{
  const float* Wp = Wp0 + k0;
  if (KFULL) {
    if (rowOk) {
      #pragma unroll
      for (int i = 0; i < 4; ++i) f[i] = Wp[i];
    } else {
      #pragma unroll
      for (int i = 0; i < 4; ++i) f[i] = 0.f;
    }
  } else {
    const int rem = din - k0 - bK;
    if (rowOk && rem >= 4) {
      #pragma unroll
      for (int i = 0; i < 4; ++i) f[i] = Wp[i];
    } else if (rowOk && rem > 0) {
      #pragma unroll
      for (int i = 0; i < 4; ++i) f[i] = (i < rem) ? Wp[i] : 0.f;
    } else {
      #pragma unroll
      for (int i = 0; i < 4; ++i) f[i] = 0.f;
    }
  }
}

__device__ __forceinline__ void pack_store16(_Float16* dst, const float* f) {
  Pack16 hb;
  #pragma unroll
  for (int i = 0; i < 16; ++i) hb.h[i] = (_Float16)f[i];
  *(uint4*)dst       = hb.q[0];
  *(uint4*)(dst + 8) = hb.q[1];
}

__device__ __forceinline__ void pack_store4(_Float16* dst, const float* f) {
  Pack4 hb;
  #pragma unroll
  for (int i = 0; i < 4; ++i) hb.h[i] = (_Float16)f[i];
  *(uint2*)dst = hb.q;
}

template <bool KFULL, bool CHECKM>
__device__ __forceinline__ void gemm_body(
    float* __restrict__ C, const float* __restrict__ A,
    const float* __restrict__ W, const float* __restrict__ bias,
    int N, int din, int dout, _Float16* As, _Float16* Bs)
{
  const int t     = threadIdx.x;
  const int mBase = blockIdx.x * TM;
  const int nBase = blockIdx.y * TN;
  const int wid   = t >> 5;
  const int lane  = t & 31;
  const int wm    = wid & 3;   // M-group of 32 rows
  const int wn    = wid >> 2;  // N subtile
  const int lm    = lane & 15;

  // Staging assignments (fixed per thread)
  const int aRow = t >> 1;          // 0..127, 16 halves each
  const int aK   = (t & 1) * 16;
  const int bRow = t >> 3;          // 0..31, 4 halves each
  const int bK   = (t & 7) * 4;

  const bool aRowOk = !CHECKM || (mBase + aRow < N);
  const bool bRowOk = (nBase + bRow < dout);
  const float* Abase = A + (size_t)(mBase + aRow) * din + aK;
  const float* Wbase = W + (size_t)(nBase + bRow) * din + bK;

  float fa[16], fb[4];
  fetch_a<KFULL>(fa, Abase, aRowOk, din, 0, aK);
  fetch_b<KFULL>(fb, Wbase, bRowOk, din, 0, bK);
  pack_store16(&As[aRow * TK + aK], fa);
  pack_store4(&Bs[bRow * TK + bK], fb);
  __syncthreads();

  v8f acc0 = {}, acc1 = {};
  const int kh = (lane < 16) ? 0 : 8;
  const int kb = (lane < 16) ? 0 : 16;
  const int r0 = wm * 32 + lm;
  const int nn = wn * 16 + lm;

  int buf = 0;
  for (int k0 = 0; k0 < din; k0 += TK) {
    const int  k1   = k0 + TK;
    const bool more = k1 < din;
    if (more) {
      __builtin_prefetch(Abase + k1 + TK, 0, 1);  // speculative next-next tile
      fetch_a<KFULL>(fa, Abase, aRowOk, din, k1, aK);
      fetch_b<KFULL>(fb, Wbase, bRowOk, din, k1, bK);
    }
    {
      const _Float16* Ab = As + buf * (TM * TK);
      const _Float16* Bb = Bs + buf * (TN * TK);
      // A frag: M=lane%16; lanes 0-15: K 0-7/16-23; lanes 16-31: K 8-15/24-31
      // B frag: N=lane%16; lanes 0-15: K 0-7/16-23; lanes 16-31: K 8-15/24-31
      FragH a0, a1, b;
      a0.q[0] = *(const uint4*)&Ab[r0 * TK + kh];
      a0.q[1] = *(const uint4*)&Ab[r0 * TK + 16 + kh];
      a1.q[0] = *(const uint4*)&Ab[(r0 + 16) * TK + kh];
      a1.q[1] = *(const uint4*)&Ab[(r0 + 16) * TK + 16 + kh];
      b.q[0]  = *(const uint4*)&Bb[nn * TK + kh];
      b.q[1]  = *(const uint4*)&Bb[nn * TK + 16 + kh];
      acc0 = __builtin_amdgcn_wmma_f32_16x16x32_f16(
          false, a0.v, false, b.v, (short)0, acc0, false, false);
      acc1 = __builtin_amdgcn_wmma_f32_16x16x32_f16(
          false, a1.v, false, b.v, (short)0, acc1, false, false);
    }
    if (more) {
      _Float16* An = As + (buf ^ 1) * (TM * TK);
      _Float16* Bn = Bs + (buf ^ 1) * (TN * TK);
      pack_store16(&An[aRow * TK + aK], fa);
      pack_store4(&Bn[bRow * TK + bK], fb);
    }
    __syncthreads();
    buf ^= 1;
  }

  // D layout: VGPR r -> (M = r or 8+r, N = lane%16)
  AccU u0, u1; u0.v = acc0; u1.v = acc1;
  const int col = nBase + wn * 16 + lm;
  if (col < dout) {
    const float bv = bias ? bias[col] : 0.f;
    const int rbase = mBase + wm * 32 + ((lane < 16) ? 0 : 8);
    #pragma unroll
    for (int r = 0; r < 8; ++r) {
      int row = rbase + r;
      if (!CHECKM || row < N) C[(size_t)row * dout + col] = u0.f[r] + bv;
    }
    #pragma unroll
    for (int r = 0; r < 8; ++r) {
      int row = rbase + 16 + r;
      if (!CHECKM || row < N) C[(size_t)row * dout + col] = u1.f[r] + bv;
    }
  }
}

template <bool KFULL>
__global__ __launch_bounds__(256) void gemm_wmma_kernel(
    float* __restrict__ C, const float* __restrict__ A,
    const float* __restrict__ W, const float* __restrict__ bias,
    int N, int din, int dout)
{
  __shared__ _Float16 As[2 * TM * TK];   // ping-pong [m][k]
  __shared__ _Float16 Bs[2 * TN * TK];   // ping-pong [n][k]
  if ((int)blockIdx.x * TM + TM <= N)
    gemm_body<KFULL, false>(C, A, W, bias, N, din, dout, As, Bs);
  else
    gemm_body<KFULL, true>(C, A, W, bias, N, din, dout, As, Bs);
}

// ---------------------------------------------------------------------------
// BatchNorm helpers (training-mode: biased stats over all rows)
// ---------------------------------------------------------------------------
__global__ void zero_kernel(float* __restrict__ p, int n) {
  int i = blockIdx.x * blockDim.x + threadIdx.x;
  if (i < n) p[i] = 0.f;
}

__global__ void colstats_kernel(float* __restrict__ stats, const float* __restrict__ Y,
                                long long N, int C, int rowsPerBlock)
{
  int c = threadIdx.x;
  if (c >= C) return;
  long long r0 = (long long)blockIdx.x * rowsPerBlock;
  long long r1 = r0 + rowsPerBlock; if (r1 > N) r1 = N;
  float s = 0.f, q = 0.f;
  for (long long r = r0; r < r1; ++r) {
    float v = Y[r * C + c];
    s += v; q += v * v;
  }
  atomicAdd(&stats[c], s);
  atomicAdd(&stats[C + c], q);
}

__global__ void bn_act_kernel(float* __restrict__ Y, const float* __restrict__ stats,
                              const float* __restrict__ g, const float* __restrict__ be,
                              long long total, int C, float invN, int act)
{
  long long i = (long long)blockIdx.x * blockDim.x + threadIdx.x;
  if (i >= total) return;
  int c = (int)(i % C);
  float m = stats[c] * invN;
  float v = stats[C + c] * invN - m * m;
  float x = (Y[i] - m) * rsqrtf(v + EPSBN) * g[c] + be[c];
  Y[i] = act_apply(x, act);
}

__global__ void add_act_kernel(float* __restrict__ out, const float* __restrict__ a,
                               const float* __restrict__ b, long long n, int act)
{
  long long i = (long long)blockIdx.x * blockDim.x + threadIdx.x;
  if (i >= n) return;
  out[i] = act_apply(a[i] + b[i], act);
}

// Softmax-attention pooling over neighbor axis:
// f[n,c] = sum_k X[n,k,c] * softmax_k(L[n,k,c])
__global__ void attpool_kernel(float* __restrict__ f, const float* __restrict__ X,
                               const float* __restrict__ L, long long N, int K, int C)
{
  long long i = (long long)blockIdx.x * blockDim.x + threadIdx.x;
  if (i >= N * (long long)C) return;
  long long n = i / C;
  int c = (int)(i % C);
  const float* Lp = L + (n * K) * (long long)C + c;
  const float* Xp = X + (n * K) * (long long)C + c;
  float mx = Lp[0];
  for (int k = 1; k < K; ++k) mx = fmaxf(mx, Lp[(long long)k * C]);
  float den = 0.f, num = 0.f;
  for (int k = 0; k < K; ++k) {
    float e = __expf(Lp[(long long)k * C] - mx);
    den += e;
    num += Xp[(long long)k * C] * e;
  }
  f[i] = num / den;
}

__global__ void rowmax_kernel(float* __restrict__ out, const float* __restrict__ X,
                              long long N, int K, int C)
{
  long long i = (long long)blockIdx.x * blockDim.x + threadIdx.x;
  if (i >= N * (long long)C) return;
  long long n = i / C;
  int c = (int)(i % C);
  const float* Xp = X + (n * K) * (long long)C + c;
  float m = Xp[0];
  for (int k = 1; k < K; ++k) m = fmaxf(m, Xp[(long long)k * C]);
  out[i] = m;
}

// ---------------------------------------------------------------------------
// Gather / concat builders
// ---------------------------------------------------------------------------
__global__ void build_nf_in_kernel(float* __restrict__ out, const float* __restrict__ pf,
                                   const int* __restrict__ cvi, const float* __restrict__ off3)
{
  int i = blockIdx.x * blockDim.x + threadIdx.x;
  if (i >= NCELL * 4) return;
  int idx = cvi[i];
  float* o = out + (size_t)i * 19;
  if (idx < 0) {
    #pragma unroll
    for (int j = 0; j < 16; ++j) o[j] = 0.f;
  } else {
    const float* p = pf + (size_t)idx * 16;
    #pragma unroll
    for (int j = 0; j < 16; ++j) o[j] = p[j];
  }
  const float* q = off3 + (size_t)i * 3;
  o[16] = q[0]; o[17] = q[1]; o[18] = q[2];
}

__global__ void build_ff_in_kernel(float* __restrict__ out, const float* __restrict__ pf,
                                   const int* __restrict__ fvi)
{
  int i = blockIdx.x * blockDim.x + threadIdx.x;
  if (i >= NFACET * 3) return;
  int idx = fvi[i];
  float* o = out + (size_t)i * 16;
  if (idx < 0) {
    #pragma unroll
    for (int j = 0; j < 16; ++j) o[j] = 0.f;
  } else {
    const float* p = pf + (size_t)idx * 16;
    #pragma unroll
    for (int j = 0; j < 16; ++j) o[j] = p[j];
  }
}

__global__ void concat_rows_kernel(float* __restrict__ out, const float* __restrict__ A,
                                   const float* __restrict__ B, long long N, int CA, int CB)
{
  long long i = (long long)blockIdx.x * blockDim.x + threadIdx.x;
  int Ct = CA + CB;
  if (i >= N * (long long)Ct) return;
  long long n = i / Ct;
  int c = (int)(i % Ct);
  out[i] = (c < CA) ? A[n * CA + c] : B[n * CB + (c - CA)];
}

// x1[NF,2,40] = [ff20 bcast | nf20[fnc]]
__global__ void build_x1_kernel(float* __restrict__ out, const float* __restrict__ ff20,
                                const float* __restrict__ nf20, const int* __restrict__ fnc)
{
  int i = blockIdx.x * blockDim.x + threadIdx.x;
  if (i >= NFACET * 2) return;
  int f = i >> 1;
  float* o = out + (size_t)i * 40;
  const float* a = ff20 + (size_t)f * 20;
  #pragma unroll
  for (int j = 0; j < 20; ++j) o[j] = a[j];
  const float* b = nf20 + (size_t)fnc[i] * 20;
  #pragma unroll
  for (int j = 0; j < 20; ++j) o[20 + j] = b[j];
}

// x2[NC,4,56] = [nf20 bcast | nf20[adj] | fetch(ff16, taf)]
__global__ void build_x2_kernel(float* __restrict__ out, const float* __restrict__ nf20,
                                const float* __restrict__ ff16, const int* __restrict__ adj,
                                const int* __restrict__ taf)
{
  int i = blockIdx.x * blockDim.x + threadIdx.x;
  if (i >= NCELL * 4) return;
  int c = i >> 2;
  float* o = out + (size_t)i * 56;
  const float* a = nf20 + (size_t)c * 20;
  #pragma unroll
  for (int j = 0; j < 20; ++j) o[j] = a[j];
  const float* b = nf20 + (size_t)adj[i] * 20;
  #pragma unroll
  for (int j = 0; j < 20; ++j) o[20 + j] = b[j];
  int fi = taf[i];
  if (fi < 0) {
    #pragma unroll
    for (int j = 0; j < 16; ++j) o[40 + j] = 0.f;
  } else {
    const float* g = ff16 + (size_t)fi * 16;
    #pragma unroll
    for (int j = 0; j < 16; ++j) o[40 + j] = g[j];
  }
}

// x3[NF,2,32] = [ff16 bcast | nf16[fnc]]
__global__ void build_x3_kernel(float* __restrict__ out, const float* __restrict__ ff16,
                                const float* __restrict__ nf16, const int* __restrict__ fnc)
{
  int i = blockIdx.x * blockDim.x + threadIdx.x;
  if (i >= NFACET * 2) return;
  int f = i >> 1;
  float* o = out + (size_t)i * 32;
  const float* a = ff16 + (size_t)f * 16;
  #pragma unroll
  for (int j = 0; j < 16; ++j) o[j] = a[j];
  const float* b = nf16 + (size_t)fnc[i] * 16;
  #pragma unroll
  for (int j = 0; j < 16; ++j) o[16 + j] = b[j];
}

// x4[NC,4,48] = [nf16 bcast | nf16[adj] | fetch(ff16b, taf)]
__global__ void build_x4_kernel(float* __restrict__ out, const float* __restrict__ nf16,
                                const float* __restrict__ ff16b, const int* __restrict__ adj,
                                const int* __restrict__ taf)
{
  int i = blockIdx.x * blockDim.x + threadIdx.x;
  if (i >= NCELL * 4) return;
  int c = i >> 2;
  float* o = out + (size_t)i * 48;
  const float* a = nf16 + (size_t)c * 16;
  #pragma unroll
  for (int j = 0; j < 16; ++j) o[j] = a[j];
  const float* b = nf16 + (size_t)adj[i] * 16;
  #pragma unroll
  for (int j = 0; j < 16; ++j) o[16 + j] = b[j];
  int fi = taf[i];
  if (fi < 0) {
    #pragma unroll
    for (int j = 0; j < 16; ++j) o[32 + j] = 0.f;
  } else {
    const float* g = ff16b + (size_t)fi * 16;
    #pragma unroll
    for (int j = 0; j < 16; ++j) o[32 + j] = g[j];
  }
}

__global__ void softmax2_kernel(float* __restrict__ out, const float* __restrict__ y, int N)
{
  int n = blockIdx.x * blockDim.x + threadIdx.x;
  if (n >= N) return;
  float a = y[2 * n], b = y[2 * n + 1];
  float m = fmaxf(a, b);
  float ea = __expf(a - m), eb = __expf(b - m);
  float d = ea + eb;
  out[2 * n]     = ea / d;
  out[2 * n + 1] = eb / d;
}

// ---------------------------------------------------------------------------
// Host-side orchestration
// ---------------------------------------------------------------------------
struct AttP { const float *Wfc, *Wm, *bem, *bm, *gm; };
struct ResP { const float *W1, *W2, *Ws, *b1, *b2, *be1, *be2, *bes, *bs, *g1, *g2, *gs; };
struct MlpP { const float *W, *b, *be, *g; };

struct Arena {
  char* base; size_t off;
  float* alloc(long long elems) {
    size_t b = ((size_t)elems * sizeof(float) + 255) & ~(size_t)255;
    float* p = (float*)(base + off); off += b; return p;
  }
};

static inline unsigned gblk(long long n, int b) { return (unsigned)((n + b - 1) / b); }

static void run_gemm(hipStream_t st, float* C, const float* A, const float* W,
                     const float* bias, long long N, int din, int dout)
{
  dim3 grid((unsigned)((N + TM - 1) / TM), (unsigned)((dout + TN - 1) / TN));
  if ((din & 31) == 0)
    gemm_wmma_kernel<true><<<grid, 256, 0, st>>>(C, A, W, bias, (int)N, din, dout);
  else
    gemm_wmma_kernel<false><<<grid, 256, 0, st>>>(C, A, W, bias, (int)N, din, dout);
}

static void run_bn(hipStream_t st, float* Y, float* stats, const float* g, const float* be,
                   long long N, int C, int act)
{
  zero_kernel<<<gblk(2 * C, 256), 256, 0, st>>>(stats, 2 * C);
  colstats_kernel<<<gblk(N, 512), 256, 0, st>>>(stats, Y, N, C, 512);
  bn_act_kernel<<<gblk(N * (long long)C, 256), 256, 0, st>>>(
      Y, stats, g, be, N * (long long)C, C, 1.f / (float)N, act);
}

static void run_res(hipStream_t st, const ResP& p, const float* X, float* out,
                    float* h1, float* h2, float* sb, float* stats,
                    long long N, int din, int dm, int dout, int act)
{
  run_gemm(st, h1, X, p.W1, p.b1, N, din, dm);
  run_bn(st, h1, stats, p.g1, p.be1, N, dm, act);
  run_gemm(st, h2, h1, p.W2, p.b2, N, dm, dout);
  run_bn(st, h2, stats, p.g2, p.be2, N, dout, ACT_NONE);
  run_gemm(st, sb, X, p.Ws, p.bs, N, din, dout);
  run_bn(st, sb, stats, p.gs, p.bes, N, dout, ACT_NONE);
  add_act_kernel<<<gblk(N * (long long)dout, 256), 256, 0, st>>>(
      out, h2, sb, N * (long long)dout, act);
}

static void run_att(hipStream_t st, const AttP& p, const float* X, float* out,
                    float* logits, float* f, float* stats,
                    long long N, int K, int C, int dout, int act, bool last)
{
  run_gemm(st, logits, X, p.Wfc, nullptr, N * K, C, C);
  attpool_kernel<<<gblk(N * (long long)C, 256), 256, 0, st>>>(f, X, logits, N, K, C);
  run_gemm(st, out, f, p.Wm, p.bm, N, C, dout);
  if (!last) run_bn(st, out, stats, p.gm, p.bem, N, dout, act);
}

static void run_mlp(hipStream_t st, const MlpP& p, const float* X, float* out, float* stats,
                    long long N, int din, int dout)
{
  run_gemm(st, out, X, p.W, p.b, N, din, dout);
  run_bn(st, out, stats, p.g, p.be, N, dout, ACT_RELU);
}

extern "C" void kernel_launch(void* const* d_in, const int* in_sizes, int n_in,
                              void* d_out, int out_size, void* d_ws, size_t ws_size,
                              hipStream_t stream)
{
  (void)in_sizes; (void)n_in; (void)out_size; (void)ws_size;

  const float* bpf   = (const float*)d_in[0];  // [NP,KP,GEO]
  const float* coxyz = (const float*)d_in[1];  // [NC,4,3]
  const float* bcf   = (const float*)d_in[2];  // [NC,4]
  const float* bff   = (const float*)d_in[3];  // [NF,4]
  const int*   cvi   = (const int*)d_in[4];    // [NC,4]
  const int*   fvi   = (const int*)d_in[5];    // [NF,3]
  const int*   fnc   = (const int*)d_in[6];    // [NF,2]
  const int*   adj   = (const int*)d_in[7];    // [NC,4]
  const int*   taf   = (const int*)d_in[8];    // [NC,4]

  // Param leaves: jax pytree, dict keys sorted (ASCII: 'W' < 'b' < 'g').
  const float* const* P = (const float* const*)(d_in + 9);
  auto att_at = [&](int i) { return AttP{P[i], P[i+1], P[i+2], P[i+3], P[i+4]}; };
  auto res_at = [&](int i) { return ResP{P[i], P[i+1], P[i+2], P[i+3], P[i+4], P[i+5],
                                         P[i+6], P[i+7], P[i+8], P[i+9], P[i+10], P[i+11]}; };
  auto mlp_at = [&](int i) { return MlpP{P[i], P[i+1], P[i+2], P[i+3]}; };

  AttP att0 = att_at(0);
  AttP att1 = att_at(5);
  AttP att2 = AttP{P[10], P[11], nullptr, P[12], nullptr};  // last=True: Wfc, Wm, bm
  AttP attp = att_at(13);
  ResP cfe1 = res_at(18);
  ResP cfe2 = res_at(30);
  ResP cne1 = res_at(42);
  ResP cne2a = res_at(54);
  ResP cne2b = res_at(66);
  MlpP m21 = mlp_at(78);
  MlpP m22 = mlp_at(82);
  MlpP m23 = mlp_at(86);
  ResP ne  = res_at(90);
  ResP pe1 = res_at(102);
  ResP pe2 = res_at(114);

  Arena ar{(char*)d_ws, 0};
  float* stats = ar.alloc(1024);                     // colsum | colsumsq (C<=256)
  float* pf    = ar.alloc((long long)NPTS * 16);     // point features
  float* nf20  = ar.alloc((long long)NCELL * 20);
  float* ff20  = ar.alloc((long long)NFACET * 20);
  float* ff16a = ar.alloc((long long)NFACET * 16);   // cycle-1 facet out
  float* nf16a = ar.alloc((long long)NCELL * 16);    // cycle-1 node out
  float* ff16b = ar.alloc((long long)NFACET * 16);   // cycle-2 facet out
  const size_t mark0 = ar.off;

  // ---- Point encoder: [NP*KP, 8] -> pe1 -> pe2 -> att pool -> pf [NP,16]
  {
    const long long R = (long long)NPTS * KPNB;
    float* pf1 = ar.alloc(R * 64);
    size_t m2 = ar.off;
    float* h1 = ar.alloc(R * 32);
    float* h2 = ar.alloc(R * 64);
    float* sb = ar.alloc(R * 64);
    run_res(stream, pe1, bpf, pf1, h1, h2, sb, stats, R, GEO, 32, 64, ACT_L02);
    ar.off = m2;
    float* pf2 = ar.alloc(R * 64);
    size_t m3 = ar.off;
    float* h1b = ar.alloc(R * 128);
    float* h2b = ar.alloc(R * 64);
    float* sb2 = ar.alloc(R * 64);
    run_res(stream, pe2, pf1, pf2, h1b, h2b, sb2, stats, R, 64, 128, 64, ACT_L02);
    ar.off = m3;
    float* logits = ar.alloc(R * 64);
    float* fbuf   = ar.alloc((long long)NPTS * 64);
    run_att(stream, attp, pf2, pf, logits, fbuf, stats, NPTS, KPNB, 64, 16, ACT_L02, false);
    ar.off = mark0;
  }

  // ---- Node (cell) features: gather + ne res + att0 -> nf20 [NC,20]
  {
    const long long Rn = (long long)NCELL * 4;
    float* nfin = ar.alloc(Rn * 19);
    build_nf_in_kernel<<<gblk(Rn, 256), 256, 0, stream>>>(nfin, pf, cvi, coxyz);
    float* resO = ar.alloc(Rn * 16);
    float* h1 = ar.alloc(Rn * 64);
    float* h2 = ar.alloc(Rn * 16);
    float* sb = ar.alloc(Rn * 16);
    run_res(stream, ne, nfin, resO, h1, h2, sb, stats, Rn, 19, 64, 16, ACT_L02);
    float* logits = ar.alloc(Rn * 16);
    float* fbuf   = ar.alloc((long long)NCELL * 16);
    float* y      = ar.alloc((long long)NCELL * 16);
    run_att(stream, att0, resO, y, logits, fbuf, stats, NCELL, 4, 16, 16, ACT_L02, false);
    concat_rows_kernel<<<gblk((long long)NCELL * 20, 256), 256, 0, stream>>>(
        nf20, y, bcf, NCELL, 16, 4);
    ar.off = mark0;
  }

  // ---- Facet features: gather + att1 -> ff20 [NF,20]
  {
    const long long Rf = (long long)NFACET * 3;
    float* ffin = ar.alloc(Rf * 16);
    build_ff_in_kernel<<<gblk(Rf, 256), 256, 0, stream>>>(ffin, pf, fvi);
    float* logits = ar.alloc(Rf * 16);
    float* fbuf   = ar.alloc((long long)NFACET * 16);
    float* y      = ar.alloc((long long)NFACET * 16);
    run_att(stream, att1, ffin, y, logits, fbuf, stats, NFACET, 3, 16, 16, ACT_L02, false);
    concat_rows_kernel<<<gblk((long long)NFACET * 20, 256), 256, 0, stream>>>(
        ff20, y, bff, NFACET, 16, 4);
    ar.off = mark0;
  }

  // ---- Cycle 1: facet update -> ff16a
  {
    const long long R2 = (long long)NFACET * 2;
    float* x1 = ar.alloc(R2 * 40);
    build_x1_kernel<<<gblk(R2, 256), 256, 0, stream>>>(x1, ff20, nf20, fnc);
    float* resO = ar.alloc(R2 * 128);
    float* h1 = ar.alloc(R2 * 64);
    float* h2 = ar.alloc(R2 * 128);
    float* sb = ar.alloc(R2 * 128);
    run_res(stream, cfe1, x1, resO, h1, h2, sb, stats, R2, 40, 64, 128, ACT_RELU);
    float* mx = ar.alloc((long long)NFACET * 128);
    rowmax_kernel<<<gblk((long long)NFACET * 128, 256), 256, 0, stream>>>(mx, resO, NFACET, 2, 128);
    run_mlp(stream, m21, mx, ff16a, stats, NFACET, 128, 16);
    ar.off = mark0;
  }

  // ---- Cycle 1: node update -> nf16a
  {
    const long long R4 = (long long)NCELL * 4;
    float* x2 = ar.alloc(R4 * 56);
    build_x2_kernel<<<gblk(R4, 256), 256, 0, stream>>>(x2, nf20, ff16a, adj, taf);
    float* resO = ar.alloc(R4 * 128);
    float* h1 = ar.alloc(R4 * 64);
    float* h2 = ar.alloc(R4 * 128);
    float* sb = ar.alloc(R4 * 128);
    run_res(stream, cne1, x2, resO, h1, h2, sb, stats, R4, 56, 64, 128, ACT_RELU);
    float* mx = ar.alloc((long long)NCELL * 128);
    rowmax_kernel<<<gblk((long long)NCELL * 128, 256), 256, 0, stream>>>(mx, resO, NCELL, 4, 128);
    run_mlp(stream, m22, mx, nf16a, stats, NCELL, 128, 16);
    ar.off = mark0;
  }

  // ---- Cycle 2: facet update -> ff16b
  {
    const long long R2 = (long long)NFACET * 2;
    float* x3 = ar.alloc(R2 * 32);
    build_x3_kernel<<<gblk(R2, 256), 256, 0, stream>>>(x3, ff16a, nf16a, fnc);
    float* resO = ar.alloc(R2 * 128);
    float* h1 = ar.alloc(R2 * 64);
    float* h2 = ar.alloc(R2 * 128);
    float* sb = ar.alloc(R2 * 128);
    run_res(stream, cfe2, x3, resO, h1, h2, sb, stats, R2, 32, 64, 128, ACT_RELU);
    float* mx = ar.alloc((long long)NFACET * 128);
    rowmax_kernel<<<gblk((long long)NFACET * 128, 256), 256, 0, stream>>>(mx, resO, NFACET, 2, 128);
    run_mlp(stream, m23, mx, ff16b, stats, NFACET, 128, 16);
    ar.off = mark0;
  }

  // ---- Cycle 2: node update -> cne2a -> cne2b -> att2 -> softmax -> d_out
  {
    const long long R4 = (long long)NCELL * 4;
    float* r1 = ar.alloc(R4 * 128);
    size_t m2 = ar.off;
    float* x4 = ar.alloc(R4 * 48);
    float* h1 = ar.alloc(R4 * 64);
    float* h2 = ar.alloc(R4 * 128);
    float* sb = ar.alloc(R4 * 128);
    build_x4_kernel<<<gblk(R4, 256), 256, 0, stream>>>(x4, nf16a, ff16b, adj, taf);
    run_res(stream, cne2a, x4, r1, h1, h2, sb, stats, R4, 48, 64, 128, ACT_L01);
    ar.off = m2;
    float* r2 = ar.alloc(R4 * 128);
    size_t m3 = ar.off;
    float* h1b = ar.alloc(R4 * 256);
    float* h2b = ar.alloc(R4 * 128);
    float* sb2 = ar.alloc(R4 * 128);
    run_res(stream, cne2b, r1, r2, h1b, h2b, sb2, stats, R4, 128, 256, 128, ACT_L01);
    ar.off = m3;
    float* logits = ar.alloc(R4 * 128);
    float* fbuf   = ar.alloc((long long)NCELL * 128);
    float* y      = ar.alloc((long long)NCELL * 2);
    run_att(stream, att2, r2, y, logits, fbuf, stats, NCELL, 4, 128, 2, ACT_NONE, true);
    softmax2_kernel<<<gblk(NCELL, 256), 256, 0, stream>>>((float*)d_out, y, NCELL);
  }
}
